// OpticalFlowEstimator_1133871366494
// MI455X (gfx1250) — compile-verified, matrix operand order checked by reading, not run on hardware
//
#include <hip/hip_runtime.h>

// ---------------------------------------------------------------------------
// Optical-flow estimator head for MI455X (gfx1250, wave32, WMMA + TDM).
//  - cost volume + concat : LDS-tiled, packed v_pk_fma_f16, f16 NHWC out
//  - conv1/2/3 (+ReLU)    : implicit-GEMM, v_wmma_f32_16x16x32_f16,
//                           halo staged by tensor_load_to_lds (TDM, TENSORcnt)
//  - conv4 (32 -> 2)      : LDS-tiled VALU kernel, f32 NCHW to d_out
// All intermediate activations use a zero-padded (H+2)x(W+2) NHWC layout so
// every halo is a clean in-bounds 2D tile (no bounds checks, TDM-friendly).
// ---------------------------------------------------------------------------

typedef __attribute__((ext_vector_type(16))) _Float16 v16h;
typedef __attribute__((ext_vector_type(8)))  _Float16 v8h;
typedef __attribute__((ext_vector_type(2)))  _Float16 h2;
typedef __attribute__((ext_vector_type(8)))  float    v8f;
typedef __attribute__((ext_vector_type(4)))  unsigned u32x4;
typedef __attribute__((ext_vector_type(8)))  unsigned u32x8;

#define BATCH 16
#define HH    128
#define WW    128
#define HP    130     // padded height (1-pixel zero border)
#define WP    130     // padded width
#define C0    64      // feature channels of f1/f2
#define C1IN  145     // 64 + 81 cost-volume channels
#define C1PAD 160     // padded to multiple of 32 (one WMMA K-chunk per tap)

// 16-bit A-matrix 16x32 per-lane K index (CDNA5 ISA 7.12.2).
static __device__ __forceinline__ int klocal(int g, int t) {
  int v = t >> 1, h = t & 1;
  return (v < 4 ? 2 * v + h : 16 + 2 * (v - 4) + h) + 8 * g;
}

// ---------------------------------------------------------------------------
// Zero the 1-pixel border of a padded NHWC f16 buffer (516 border pixels per
// image). Each thread writes 8 halves (16B).
// ---------------------------------------------------------------------------
template <int CPAD>
__global__ void zero_border(_Float16* __restrict__ buf) {
  constexpr int CW = CPAD / 8;
  int tid = blockIdx.x * 256 + threadIdx.x;
  if (tid >= BATCH * 516 * CW) return;
  int cw = tid % CW;
  int p  = (tid / CW) % 516;
  int b  = tid / (516 * CW);
  int y, x;
  if (p < 130)      { y = 0;   x = p; }
  else if (p < 260) { y = 129; x = p - 130; }
  else { int q = p - 260; y = 1 + (q >> 1); x = (q & 1) ? 129 : 0; }
  v8h z = {};
  *(v8h*)(buf + (((size_t)b * HP + y) * WP + x) * CPAD + cw * 8) = z;
}

// ---------------------------------------------------------------------------
// Repack f32 OIHW conv weights into f16 WMMA A-fragments:
//   wf[((coTile*nK + kc)*32 + lane)*16 + t]
// ---------------------------------------------------------------------------
template <int CIN, int CPAD, int COUT>
__global__ void repack_weights(const float* __restrict__ w,
                               _Float16* __restrict__ wf) {
  constexpr int nK = 9 * CPAD / 32;
  constexpr int total = (COUT / 16) * nK * 512;
  int tid = blockIdx.x * 256 + threadIdx.x;
  if (tid >= total) return;
  int t    = tid & 15;
  int lane = (tid >> 4) & 31;
  int kc   = (tid >> 9) % nK;
  int coT  = tid / (nK * 512);
  int g = lane >> 4, m = lane & 15;
  int co = coT * 16 + m;
  int K  = kc * 32 + klocal(g, t);
  int tap = K / CPAD;
  int ci  = K % CPAD;
  float val = (ci < CIN) ? w[(co * CIN + ci) * 9 + tap] : 0.0f;
  wf[tid] = (_Float16)val;
}

// ---------------------------------------------------------------------------
// Cost volume + concat, LDS tiled. Block = 16x16 output pixels.
// Per channel-pair: stage a 24x24 packed-f16 f2 tile in LDS, then every
// thread runs 81 v_pk_fma_f16 accumulations. f1 is written through as f16.
// Output: padded NHWC [B,130,130,160] (interior only; borders pre-zeroed).
// ---------------------------------------------------------------------------
__global__ void costvol_concat(const float* __restrict__ f1,
                               const float* __restrict__ f2,
                               _Float16* __restrict__ x0) {
  __shared__ alignas(16) h2 t2[24 * 24];
  int xt = blockIdx.x & 7;
  int yt = (blockIdx.x >> 3) & 7;
  int b  = blockIdx.x >> 6;
  int y0 = yt * 16, x0c = xt * 16;
  int tx = threadIdx.x & 15, ty = threadIdx.x >> 4;
  int y = y0 + ty, x = x0c + tx;

  _Float16* outp = x0 + (((size_t)b * HP + y + 1) * WP + (x + 1)) * C1PAD;
  const float* f1b = f1 + (size_t)b * C0 * HH * WW;
  const float* f2b = f2 + (size_t)b * C0 * HH * WW;

  h2 acc[81];
#pragma unroll
  for (int i = 0; i < 81; ++i) acc[i] = h2{};

  for (int cp = 0; cp < 32; ++cp) {
    const float* p0 = f2b + (size_t)(2 * cp) * HH * WW;
    const float* p1 = f2b + (size_t)(2 * cp + 1) * HH * WW;
    for (int i = threadIdx.x; i < 576; i += 256) {
      int r = i / 24, c = i - r * 24;
      int yy = y0 - 4 + r, xx = x0c - 4 + c;
      h2 v = {};
      if (yy >= 0 && yy < HH && xx >= 0 && xx < WW) {
        size_t o = (size_t)yy * WW + xx;
        v[0] = (_Float16)p0[o];
        v[1] = (_Float16)p1[o];
      }
      t2[i] = v;
    }
    __syncthreads();

    size_t fo = (size_t)y * WW + x;
    h2 av;
    av[0] = (_Float16)f1b[(size_t)(2 * cp) * HH * WW + fo];
    av[1] = (_Float16)f1b[(size_t)(2 * cp + 1) * HH * WW + fo];
    *(h2*)(outp + 2 * cp) = av;  // write-through f1 channels (f16)

#pragma unroll
    for (int ii = 0; ii < 9; ++ii) {
#pragma unroll
      for (int jj = 0; jj < 9; ++jj) {
        h2 bv = t2[(ty + 8 - ii) * 24 + (tx + 8 - jj)];
        acc[ii * 9 + jj] += av * bv;   // v_pk_fma_f16
      }
    }
    __syncthreads();
  }

#pragma unroll
  for (int idx = 0; idx < 81; ++idx) {
    float s = (float)acc[idx][0] + (float)acc[idx][1];
    outp[C0 + idx] = (_Float16)(s * (1.0f / 64.0f));
  }
#pragma unroll
  for (int c = C1IN; c < C1PAD; ++c) outp[c] = (_Float16)0.0f;
}

// ---------------------------------------------------------------------------
// 3x3 SAME conv + bias (+ReLU) as implicit GEMM with WMMA f16->f32.
// One block per 16-pixel row tile; COUT/16 waves. The 3x18xCPAD f16 halo is
// DMA'd into LDS with one tensor_load_to_lds (wave 0 issues; TDM ignores
// EXEC) and synchronized with s_wait_tensorcnt + barrier.
// ---------------------------------------------------------------------------
template <int CPAD, int COUT, bool RELU>
__global__ void conv3x3_wmma(const _Float16* __restrict__ xin,
                             const _Float16* __restrict__ wf,
                             const float* __restrict__ bias,
                             _Float16* __restrict__ xout) {
  constexpr int nK = 9 * CPAD / 32;
  __shared__ alignas(16) _Float16 halo[3 * 18 * CPAD];  // only LDS var -> offset 0

  int xt = blockIdx.x & 7;            // W/16 = 8 tiles per row
  int y  = (blockIdx.x >> 3) & (HH - 1);
  int b  = blockIdx.x >> 10;
  int x0 = xt * 16;

  if (threadIdx.x < 32) {  // wave 0 only (wave-uniform guard; TDM ignores EXEC)
    unsigned long long gaddr =
        (unsigned long long)(const void*)(xin + (((size_t)b * HP + y) * WP + x0) * CPAD);
    // D# group 0: count=1, lds_addr=0 (halo at LDS offset 0), global addr, type=2.
    u32x4 g0;
    g0[0] = 1u;
    g0[1] = 0u;
    g0[2] = (unsigned)gaddr;
    g0[3] = ((unsigned)(gaddr >> 32) & 0x01FFFFFFu) | 0x80000000u;
    // D# group 1: data_size=2B, tile = 3 rows x (18*CPAD) elems, row stride
    // WP*CPAD elems, tensor dims huge (tile is always in-bounds, no clipping).
    u32x8 g1;
    g1[0] = 0x00010000u;                       // data_size=1 (2 bytes)
    g1[1] = 0u;                                // tensor_dim0[15:0]=0 (dim0=0x40000000)
    g1[2] = 0x4000u;                           // dim0 hi | dim1 lo
    g1[3] = 0x4000u | ((18u * CPAD) << 16);    // dim1 hi | tile_dim0
    g1[4] = 3u;                                // tile_dim1=3, tile_dim2=0
    g1[5] = (unsigned)(WP * CPAD);             // tensor_dim0_stride lo
    g1[6] = 0u;
    g1[7] = 0u;
    asm volatile("tensor_load_to_lds %0, %1" :: "s"(g0), "s"(g1) : "memory");
    __builtin_amdgcn_s_wait_tensorcnt(0);
  }
  __syncthreads();

  int wave = threadIdx.x >> 5;
  int lane = threadIdx.x & 31;
  int n = lane & 15;     // pixel within tile (B/D column)
  int g = lane >> 4;     // K half-select group

  v8f acc = {};
  const v16h* afrag = (const v16h*)wf + (size_t)wave * nK * 32 + lane;

#pragma unroll 2
  for (int kc = 0; kc < nK; ++kc) {
    int K0  = kc * 32;
    int tap = K0 / CPAD;
    int c0  = K0 - tap * CPAD;
    int r = tap / 3;
    int s = tap - 3 * r;
    int px = n + s;
    const _Float16* bp = halo + ((r * 18 + px) * CPAD + c0 + 8 * g);
    union { v16h v; v8h h[2]; } bf;
    bf.h[0] = *(const v8h*)bp;         // K = c0 + 8g .. +7
    bf.h[1] = *(const v8h*)(bp + 16);  // K = c0 + 16 + 8g .. +7
    v16h af = afrag[(size_t)kc * 32];
    acc = __builtin_amdgcn_wmma_f32_16x16x32_f16(
        false, af, false, bf.v, (short)0, acc, false, false);
  }

  // D layout: lane holds 8 consecutive output channels for pixel n.
  int cobase = wave * 16 + 8 * g;
  v8h outv;
#pragma unroll
  for (int rr = 0; rr < 8; ++rr) {
    float v = acc[rr] + bias[cobase + rr];
    if (RELU) v = fmaxf(v, 0.0f);
    outv[rr] = (_Float16)v;
  }
  *(v8h*)(xout + (((size_t)b * HP + y + 1) * WP + (x0 + n + 1)) * COUT + cobase) = outv;
}

// ---------------------------------------------------------------------------
// Final 32 -> 2 conv (no ReLU), LDS-tiled (padded input => branch-free halo),
// f32 NCHW output. Block = 16x16 output pixels.
// ---------------------------------------------------------------------------
__global__ void conv4_flow(const _Float16* __restrict__ x3,
                           const float* __restrict__ w4,
                           const float* __restrict__ b4,
                           float* __restrict__ out) {
  __shared__ alignas(16) _Float16 tile[18 * 18 * 32];
  __shared__ float wsm[576];
  __shared__ float bsm[2];

  int xt = blockIdx.x & 7;
  int yt = (blockIdx.x >> 3) & 7;
  int b  = blockIdx.x >> 6;
  int y0 = yt * 16, x0 = xt * 16;

  // Stage halo: padded rows y0..y0+17, cols x0..x0+17, 32 ch (contiguous per row).
  constexpr int wordsPerRow = 18 * 32 / 2;  // 288 u32 words
  for (int wd = threadIdx.x; wd < 18 * wordsPerRow; wd += 256) {
    int row = wd / wordsPerRow;
    int rw  = wd - row * wordsPerRow;
    const unsigned* src =
        (const unsigned*)(x3 + (((size_t)b * HP + y0 + row) * WP + x0) * 32);
    ((unsigned*)tile)[wd] = src[rw];
  }
  for (int i = threadIdx.x; i < 576; i += 256) wsm[i] = w4[i];
  if (threadIdx.x < 2) bsm[threadIdx.x] = b4[threadIdx.x];
  __syncthreads();

  int tx = threadIdx.x & 15, ty = threadIdx.x >> 4;
  float acc0 = bsm[0], acc1 = bsm[1];
#pragma unroll
  for (int r = 0; r < 3; ++r) {
#pragma unroll
    for (int s = 0; s < 3; ++s) {
      int tap = r * 3 + s;
      const _Float16* xp = tile + (((ty + r) * 18) + tx + s) * 32;
#pragma unroll
      for (int ci = 0; ci < 32; ++ci) {
        float xv = (float)xp[ci];
        acc0 += xv * wsm[ci * 9 + tap];
        acc1 += xv * wsm[288 + ci * 9 + tap];
      }
    }
  }
  size_t o = ((size_t)b * 2 * HH + (y0 + ty)) * WW + x0 + tx;
  out[o] = acc0;
  out[o + (size_t)HH * WW] = acc1;
}

// ---------------------------------------------------------------------------
extern "C" void kernel_launch(void* const* d_in, const int* in_sizes, int n_in,
                              void* d_out, int out_size, void* d_ws, size_t ws_size,
                              hipStream_t stream) {
  const float* f1 = (const float*)d_in[0];
  const float* f2 = (const float*)d_in[1];
  const float* w1 = (const float*)d_in[2];
  const float* b1 = (const float*)d_in[3];
  const float* w2 = (const float*)d_in[4];
  const float* b2 = (const float*)d_in[5];
  const float* w3 = (const float*)d_in[6];
  const float* b3 = (const float*)d_in[7];
  const float* w4 = (const float*)d_in[8];
  const float* b4 = (const float*)d_in[9];
  float* out = (float*)d_out;

  char* ws = (char*)d_ws;
  const size_t X0_BYTES = (size_t)BATCH * HP * WP * C1PAD * 2;  // 86,528,000
  const size_t X1_BYTES = (size_t)BATCH * HP * WP * 128 * 2;    // 69,222,400
  _Float16* x0 = (_Float16*)ws;               // [B,130,130,160] f16
  _Float16* x1 = (_Float16*)(ws + X0_BYTES);  // [B,130,130,128] f16
  _Float16* x2 = (_Float16*)ws;               // [B,130,130,64], aliases dead x0
  _Float16* x3 = (_Float16*)(ws + X0_BYTES);  // [B,130,130,32], aliases dead x1
  char* wfb = ws + X0_BYTES + X1_BYTES;
  const size_t WF1 = (size_t)8 * 45 * 512 * 2;
  const size_t WF2 = (size_t)4 * 36 * 512 * 2;
  _Float16* wf1 = (_Float16*)wfb;
  _Float16* wf2 = (_Float16*)(wfb + WF1);
  _Float16* wf3 = (_Float16*)(wfb + WF1 + WF2);

  // Weight repack (small, independent).
  repack_weights<145, 160, 128><<<(8 * 45 * 512 + 255) / 256, 256, 0, stream>>>(w1, wf1);
  repack_weights<128, 128,  64><<<(4 * 36 * 512 + 255) / 256, 256, 0, stream>>>(w2, wf2);
  repack_weights< 64,  64,  32><<<(2 * 18 * 512 + 255) / 256, 256, 0, stream>>>(w3, wf3);

  // Zero-pad borders of x0/x1 (distinct regions, safe up-front).
  zero_border<160><<<(BATCH * 516 * 20 + 255) / 256, 256, 0, stream>>>(x0);
  zero_border<128><<<(BATCH * 516 * 16 + 255) / 256, 256, 0, stream>>>(x1);

  // Cost volume + concat -> x0 interior.
  costvol_concat<<<BATCH * 8 * 8, 256, 0, stream>>>(f1, f2, x0);

  const int nTiles = BATCH * HH * (WW / 16);  // 16384
  conv3x3_wmma<160, 128, true><<<nTiles, 256, 0, stream>>>(x0, wf1, b1, x1);

  zero_border<64><<<(BATCH * 516 * 8 + 255) / 256, 256, 0, stream>>>(x2);  // x0 dead
  conv3x3_wmma<128,  64, true><<<nTiles, 128, 0, stream>>>(x1, wf2, b2, x2);

  zero_border<32><<<(BATCH * 516 * 4 + 255) / 256, 256, 0, stream>>>(x3);  // x1 dead
  conv3x3_wmma< 64,  32, true><<<nTiles,  64, 0, stream>>>(x2, wf3, b3, x3);

  // Final flow head -> d_out (f32 NCHW [16,2,128,128]).
  conv4_flow<<<BATCH * 8 * 8, 256, 0, stream>>>(x3, w4, b4, out);
}
